// CCAMemoryModel_12610023981073
// MI455X (gfx1250) — matrix operand order repaired and checked by reading, not compiled, not run on hardware
//
#include <hip/hip_runtime.h>
#include <hip/hip_bf16.h>
#include <stdint.h>

// ---------------------------------------------------------------------------
// CCA memory model for MI455X (gfx1250, wave32).
//  * ~27 GFLOP total, ~15MB inputs -> compute/latency bound, not HBM-bound.
//  * attn_batch in the reference is dead code -> skipped.
//  * Each CCA block (conv_base ∘ conv_mem, linear in between) folds into ONE
//    81-tap 4D conv with combined weights -> removes 327MB t1/t2 tensors.
//  * bn chunked by 64 so block-1 output h (~23MB bf16) is L2-resident (192MB)
//    between the two CCA kernels.
//  * Matrix work on v_wmma_f32_16x16x32_bf16 (f32 accumulate):
//      - corr einsum: 512 GEMMs of 100x100x64 (M,N padded to 112)
//      - CCA block 1: im2col GEMM M=16(out ch), N=100(base pos), K=81->96
//    Outputs stored to PADDED buffers -> all WMMA store paths unconditional.
//    Zero-padded A-weights annihilate K-pad taps -> branch-free B gathers.
//  * k_corr operand staging uses global_load_async_to_lds_b128 (ASYNCcnt) via
//    inline asm; LDS rows stride 144B (16B-aligned, bank-conflict-free).
//  * Scratch requirement ~52 MB in d_ws.
// ---------------------------------------------------------------------------

#ifndef USE_ASYNC_COPY
#define USE_ASYNC_COPY 1
#endif

typedef __attribute__((ext_vector_type(16))) __bf16 v16bf;
typedef __attribute__((ext_vector_type(8)))  float  v8f;

#define TEMP_ATTN 5.0f
#define BN_EPS 1e-5f
#define L2_EPS 1e-8f

#define CP 112        // padded spatial extent (ij / hw)
#define CPP (112*112) // padded per-bn corr plane

// ---------------------------------------------------------------- utilities
__global__ __launch_bounds__(256) void k_zero(unsigned* __restrict__ p, int n) {
    int i = blockIdx.x * 256 + threadIdx.x;
    if (i < n) p[i] = 0u;
}

// Combined CCA weights + row-sums of w1x1 (for the channel-mean shift).
__global__ __launch_bounds__(256) void k_weights(
    const float* __restrict__ w1x1, const float* __restrict__ w1a,
    const float* __restrict__ w1b,  const float* __restrict__ w2a,
    const float* __restrict__ w2b,
    float* __restrict__ wsum, __bf16* __restrict__ W1bf, float* __restrict__ W2) {
    int t = threadIdx.x;
    for (int r = t; r < 64; r += 256) {
        float s = 0.f;
        for (int c = 0; c < 512; ++c) s += w1x1[r * 512 + c];
        wsum[r] = s;
    }
    // W1[o][k], k=(a*9+b), zero-padded to 96 taps (pads kill junk B operands).
    for (int idx = t; idx < 16 * 96; idx += 256) {
        int o = idx / 96, k = idx % 96;
        float v = 0.f;
        if (k < 81) {
            int a = k / 9, b = k % 9;
            for (int i = 0; i < 16; ++i)
                v += w1b[(o * 16 + i) * 9 + a] * w1a[i * 9 + b];
        }
        W1bf[idx] = (__bf16)v;
    }
    // W2[i][k] = w2b[a] * w2a[i][b]
    for (int idx = t; idx < 16 * 81; idx += 256) {
        int i = idx / 81, k = idx % 81;
        W2[idx] = w2b[k / 9] * w2a[i * 9 + (k % 9)];
    }
}

// ------------------------------------------------- projection (1x1+BN+ReLU+L2)
// One block per spatial position; 64 threads (one per output channel r).
// Output layout [outer][pos(112, zero-padded)][r(64)] so k_corr staging is a
// straight row copy (async-to-LDS friendly).
__global__ __launch_bounds__(64) void k_proj(
    const float* __restrict__ src, const float* __restrict__ w1x1,
    const float* __restrict__ wsum, const float* __restrict__ gamma,
    const float* __restrict__ beta, __bf16* __restrict__ dst) {
    __shared__ float xs[512];
    __shared__ float red[64];
    int outer = blockIdx.x / 100, hw = blockIdx.x % 100;
    int t = threadIdx.x;
    const float* x = src + (size_t)outer * 512 * 100 + hw;   // channel stride 100
    float psum = 0.f;
    for (int j = 0; j < 8; ++j) {
        float v = x[(size_t)(t + 64 * j) * 100];
        xs[t + 64 * j] = v;
        psum += v;
    }
    red[t] = psum; __syncthreads();
    for (int s2 = 32; s2 > 0; s2 >>= 1) { if (t < s2) red[t] += red[t + s2]; __syncthreads(); }
    float mu = red[0] * (1.f / 512.f);
    __syncthreads();   // red reused below
    float acc = 0.f;
    const float* wr = w1x1 + (size_t)t * 512;
    for (int c = 0; c < 512; ++c) acc += xs[c] * wr[c];
    float y = fmaxf(gamma[t] * (acc - mu * wsum[t]) + beta[t], 0.f);
    red[t] = y * y; __syncthreads();
    for (int s2 = 32; s2 > 0; s2 >>= 1) { if (t < s2) red[t] += red[t + s2]; __syncthreads(); }
    float nrm = fmaxf(sqrtf(red[0]), L2_EPS);
    dst[((size_t)outer * CP + hw) * 64 + t] = (__bf16)(y / nrm);
}

// ----------------------------------------------------------- corr (WMMA GEMM)
// corr[b,n,ij,hw] = sum_r fmp[n,r,ij]*fbp[b,r,hw]: M=ij, N=hw (112 padded), K=64.
// Wave w owns row-band mt=w (A fragments register-resident), sweeps 7 nt tiles.
// Operand staging via async global->LDS B128 copies (ASYNCcnt).
__global__ __launch_bounds__(256) void k_corr(
    const __bf16* __restrict__ fmp, const __bf16* __restrict__ fbp,
    float* __restrict__ corr) {
    __shared__ __align__(16) __bf16 As[112 * 72];  // [pos(ij)][k], 144B rows
    __shared__ __align__(16) __bf16 Bs[112 * 72];  // [pos(hw)][k]
    int bn = blockIdx.x;
    int b = bn >> 6, n = bn & 63;
    int t = threadIdx.x;
    const char* gA = (const char*)(fmp + (size_t)n * CP * 64);
    const char* gB = (const char*)(fbp + (size_t)b * CP * 64);
#if USE_ASYNC_COPY
    for (int i = t; i < 112 * 8; i += 256) {
        int pos = i >> 3, ch = i & 7;
        unsigned la = (unsigned)(uintptr_t)((char*)As + pos * 144 + ch * 16);
        unsigned lb = (unsigned)(uintptr_t)((char*)Bs + pos * 144 + ch * 16);
        unsigned long long ga = (unsigned long long)(uintptr_t)(gA + pos * 128 + ch * 16);
        unsigned long long gb = (unsigned long long)(uintptr_t)(gB + pos * 128 + ch * 16);
        asm volatile("global_load_async_to_lds_b128 %0, %1, off"
                     :: "v"(la), "v"(ga) : "memory");
        asm volatile("global_load_async_to_lds_b128 %0, %1, off"
                     :: "v"(lb), "v"(gb) : "memory");
    }
    asm volatile("s_wait_asynccnt 0" ::: "memory");
#else
    for (int i = t; i < 112 * 8; i += 256) {
        int pos = i >> 3, ch = i & 7;
        *(uint4*)((char*)As + pos * 144 + ch * 16) =
            *(const uint4*)(gA + pos * 128 + ch * 16);
        *(uint4*)((char*)Bs + pos * 144 + ch * 16) =
            *(const uint4*)(gB + pos * 128 + ch * 16);
    }
#endif
    __syncthreads();
    int wave = t >> 5, lane = t & 31;
    int lm = lane & 15;
    int khA = (lane < 16) ? 0 : 8;   // A: K 0-7/16-23 vs 8-15/24-31 per lane half
    int khB = (lane < 16) ? 0 : 8;   // B: K 0-7/16-23 vs 8-15/24-31 per lane half
    if (wave < 7) {
        int mt = wave;
        v16bf afr[2];
#pragma unroll
        for (int kc = 0; kc < 2; ++kc)
#pragma unroll
            for (int j = 0; j < 16; ++j) {
                int p = j >> 1, hh = j & 1;
                int kk = kc * 32 + ((p & 4) ? 16 : 0) + (p & 3) * 2 + hh + khA;
                afr[kc][j] = As[(mt * 16 + lm) * 72 + kk];
            }
        float* out = corr + (size_t)bn * CPP;
#pragma unroll
        for (int nt = 0; nt < 7; ++nt) {
            v8f c = {0.f, 0.f, 0.f, 0.f, 0.f, 0.f, 0.f, 0.f};
#pragma unroll
            for (int kc = 0; kc < 2; ++kc) {
                v16bf bb;
#pragma unroll
                for (int j = 0; j < 16; ++j)
                    bb[j] = Bs[(nt * 16 + lm) * 72 + kc * 32 + j + (j & 8) + khB];
                c = __builtin_amdgcn_wmma_f32_16x16x32_bf16(false, afr[kc], false, bb,
                                                            (short)0, c, false, false);
            }
            int nn = nt * 16 + lm;
#pragma unroll
            for (int v = 0; v < 8; ++v) {      // unconditional: padded layout
                int m = mt * 16 + v + ((lane < 16) ? 0 : 8);
                out[m * CP + nn] = c[v];
            }
        }
    }
}

// ---------------------------------------------- CCA block 1 (WMMA, combined W1)
// h[bn,o,m,s] = relu( sum_{a,b} corr[bn, m+a, s+b] * W1[o,a,b] )
// Per (bn, m): GEMM  M=16 (o), N=100 (s, 7 tiles), K=81->96 (3 chunks of 32).
// Branch-free B gather: zero-padded A weights kill K-pad taps; s>=100 columns
// produce junk that lands only in h's pad columns (never read).
__global__ __launch_bounds__(256) void k_cca1(
    const float* __restrict__ corr, const __bf16* __restrict__ W1bf,
    __bf16* __restrict__ h, int bnBase) {
    __shared__ __bf16 cs[144 * 144];  // zero-padded [mem 12x12][base 12x12]
    int bnl = blockIdx.x;
    int bn = bnBase + bnl;
    int t = threadIdx.x;
    for (int i = t; i < 144 * 144; i += 256) cs[i] = (__bf16)0.f;
    __syncthreads();
    const float* src = corr + (size_t)bn * CPP;
    for (int i = t; i < 10000; i += 256) {
        int m = i / 100, s = i % 100;
        cs[((m / 10 + 1) * 12 + m % 10 + 1) * 144 + (s / 10 + 1) * 12 + s % 10 + 1] =
            (__bf16)src[m * CP + s];
    }
    __syncthreads();
    int wave = t >> 5, lane = t & 31;
    int lm = lane & 15;
    int khA = (lane < 16) ? 0 : 8;
    int khB = (lane < 16) ? 0 : 8;
    // A fragments (combined weights) are loop-invariant -> registers.
    v16bf afr[3];
#pragma unroll
    for (int kc = 0; kc < 3; ++kc)
#pragma unroll
        for (int j = 0; j < 16; ++j) {
            int p = j >> 1, hh = j & 1;
            int kk = kc * 32 + ((p & 4) ? 16 : 0) + (p & 3) * 2 + hh + khA;
            afr[kc][j] = W1bf[lm * 96 + kk];
        }
    // Per-lane B tap offsets (relative to mem-base + base-base).
    int rel[3][16];
#pragma unroll
    for (int kc = 0; kc < 3; ++kc)
#pragma unroll
        for (int j = 0; j < 16; ++j) {
            int kk = kc * 32 + j + (j & 8) + khB;
            int kkc = kk < 81 ? kk : 80;            // clamped; A weight is 0 there
            int a = kkc / 9, b = kkc % 9;
            rel[kc][j] = ((a / 3 - 1) * 12 + (a % 3 - 1)) * 144 +
                         (b / 3 - 1) * 12 + (b % 3 - 1);
        }
    int s0 = lm;
    __bf16* hout = h + (size_t)bnl * 16 * 100 * CP;
    for (int m = wave; m < 100; m += 8) {
        int mbase = ((m / 10 + 1) * 12 + m % 10 + 1) * 144;
#pragma unroll
        for (int nt = 0; nt < 7; ++nt) {
            int s = nt * 16 + s0;
            int sc = s < 100 ? s : 99;              // clamp: junk -> pad columns
            int sbase = (sc / 10 + 1) * 12 + sc % 10 + 1;
            int base = mbase + sbase;
            v8f c = {0.f, 0.f, 0.f, 0.f, 0.f, 0.f, 0.f, 0.f};
#pragma unroll
            for (int kc = 0; kc < 3; ++kc) {
                v16bf bb;
#pragma unroll
                for (int j = 0; j < 16; ++j) bb[j] = cs[base + rel[kc][j]];
                c = __builtin_amdgcn_wmma_f32_16x16x32_bf16(false, afr[kc], false, bb,
                                                            (short)0, c, false, false);
            }
#pragma unroll
            for (int v = 0; v < 8; ++v) {           // unconditional: padded layout
                int o = v + ((lane < 16) ? 0 : 8);
                hout[((size_t)o * 100 + m) * CP + s] = (__bf16)fmaxf(c[v], 0.f);
            }
        }
    }
}

// --------------------------------------- CCA block 2 (VALU, 1 out ch, combined W2)
// out[bn,m,s] = sum_{i,a,b} h[bn,i,m+a,s+b] * W2[i,a,b]  (overwrites corr in place)
__global__ __launch_bounds__(256) void k_cca2(
    const __bf16* __restrict__ h, const float* __restrict__ W2,
    float* __restrict__ corr, int bnBase) {
    __shared__ __bf16 hs[3 * 12 * 144];  // [dy 0..2][wm pad 12][base pad 144]
    __shared__ float w2s[81];
    int blk = blockIdx.x;
    int bnl = blk / 10, hm = blk % 10;
    int bn = bnBase + bnl;
    int t = threadIdx.x;
    float acc[4] = {0.f, 0.f, 0.f, 0.f};
    int oid[4], obase[4];
#pragma unroll
    for (int o = 0; o < 4; ++o) {
        int idx = t + o * 256;
        oid[o] = idx;
        int ic = idx < 1000 ? idx : 999;
        int wm = ic / 100, s = ic % 100;
        obase[o] = (12 + wm + 1) * 144 + (s / 10 + 1) * 12 + (s % 10 + 1);
    }
    const __bf16* hc = h + (size_t)bnl * 16 * 100 * CP;
    for (int i = 0; i < 16; ++i) {
        __syncthreads();
        for (int k = t; k < 3 * 12 * 144; k += 256) hs[k] = (__bf16)0.f;
        if (t < 81) w2s[t] = W2[i * 81 + t];
        __syncthreads();
        for (int k = t; k < 3000; k += 256) {
            int dy = k / 1000, r = k % 1000, wm = r / 100, s = r % 100;
            int hmm = hm + dy - 1;
            if (hmm >= 0 && hmm < 10)
                hs[(dy * 12 + wm + 1) * 144 + (s / 10 + 1) * 12 + s % 10 + 1] =
                    hc[((size_t)i * 100 + hmm * 10 + wm) * CP + s];
        }
        __syncthreads();
#pragma unroll
        for (int a = 0; a < 9; ++a) {
#pragma unroll
            for (int b = 0; b < 9; ++b) {
                float w = w2s[a * 9 + b];
                const int rel = ((a / 3 - 1) * 12 + (a % 3 - 1)) * 144 +
                                (b / 3 - 1) * 12 + (b % 3 - 1);
#pragma unroll
                for (int o = 0; o < 4; ++o)
                    acc[o] += w * (float)hs[obase[o] + rel];
            }
        }
    }
    float* out = corr + (size_t)bn * CPP;
#pragma unroll
    for (int o = 0; o < 4; ++o)
        if (oid[o] < 1000) {   // guard needed: pad rows overlap neighbor blocks
            int wm = oid[o] / 100, s = oid[o] % 100;
            out[(hm * 10 + wm) * CP + s] = acc[o];
        }
}

// ---------------------- memory-direction gaussian-normalize + softmax + sum
// Per (b,n): stats over m per base column s, then attn[m] = sum_s softmax_m[m,s].
// (attn_batch in the reference is dead code and skipped.)
__global__ __launch_bounds__(128) void k_attn(
    const float* __restrict__ corr, float* __restrict__ attn) {
    __shared__ float xs[10000];
    __shared__ float mu[100], rs[100], zm[100], se[100];
    int bn = blockIdx.x, t = threadIdx.x;
    const float* src = corr + (size_t)bn * CPP;
    for (int i = t; i < 10000; i += 128) {
        int m = i / 100, s = i - m * 100;
        xs[i] = src[m * CP + s];
    }
    __syncthreads();
    if (t < 100) {
        int s = t;
        float sum = 0.f, sq = 0.f;
        for (int m = 0; m < 100; ++m) { float v = xs[m * 100 + s]; sum += v; sq += v * v; }
        float mean = sum * 0.01f;
        float var = fmaxf(sq - sum * mean, 0.f) * (1.f / 99.f);   // unbiased
        float r = 1.f / (sqrtf(var + BN_EPS) * TEMP_ATTN);
        float mx = -1e30f;
        for (int m = 0; m < 100; ++m) mx = fmaxf(mx, (xs[m * 100 + s] - mean) * r);
        float es = 0.f;
        for (int m = 0; m < 100; ++m) es += expf((xs[m * 100 + s] - mean) * r - mx);
        mu[s] = mean; rs[s] = r; zm[s] = mx; se[s] = es;
    }
    __syncthreads();
    if (t < 100) {
        int m = t;
        float a = 0.f;
        for (int s = 0; s < 100; ++s)
            a += expf((xs[m * 100 + s] - mu[s]) * rs[s] - zm[s]) / se[s];
        attn[(size_t)bn * 100 + m] = a;
    }
}

// ------------------------------------- attended memory embedding [B,N,C]
__global__ __launch_bounds__(256) void k_attmem(
    const float* __restrict__ attn, const float* __restrict__ feat_m,
    float* __restrict__ attm) {
    __shared__ float as_[100];
    int bn = blockIdx.x, n = bn & 63, t = threadIdx.x;
    if (t < 100) as_[t] = attn[(size_t)bn * 100 + t];
    __syncthreads();
    for (int c = t; c < 512; c += 256) {
        const float* fm = feat_m + (size_t)n * 512 * 100 + (size_t)c * 100;
        float a = 0.f;
        for (int m = 0; m < 100; ++m) a += as_[m] * fm[m];
        attm[(size_t)bn * 512 + c] = a * 0.01f;
    }
}

// --------------------------------------------------- memory-wrap head -> logits
__global__ __launch_bounds__(256) void k_head(
    const float* __restrict__ feat_b, const float* __restrict__ attm,
    const float* __restrict__ w_fc, const float* __restrict__ b_fc,
    float* __restrict__ out) {
    __shared__ float enc[512], mv[512], wl[64], red[256];
    __shared__ float einv;
    int b = blockIdx.x, t = threadIdx.x;
    for (int c = t; c < 512; c += 256) {
        const float* fb = feat_b + (size_t)b * 512 * 100 + (size_t)c * 100;
        float s = 0.f;
        for (int m = 0; m < 100; ++m) s += fb[m];
        enc[c] = s * 0.01f;
    }
    __syncthreads();
    float p = 0.f;
    for (int c = t; c < 512; c += 256) p += enc[c] * enc[c];
    red[t] = p; __syncthreads();
    for (int s2 = 128; s2 > 0; s2 >>= 1) { if (t < s2) red[t] += red[t + s2]; __syncthreads(); }
    if (t == 0) einv = 1.f / fmaxf(sqrtf(red[0]), L2_EPS);
    __syncthreads();
    if (t < 64) {
        const float* am = attm + ((size_t)b * 64 + t) * 512;
        float d = 0.f, nr = 0.f;
        for (int c = 0; c < 512; ++c) { float v = am[c]; d += enc[c] * v; nr += v * v; }
        wl[t] = d * einv / fmaxf(sqrtf(nr), L2_EPS);
    }
    __syncthreads();
    if (t == 0) {
        float mx = wl[0];
        for (int n = 1; n < 64; ++n) mx = fmaxf(mx, wl[n]);
        float s = 0.f;
        for (int n = 0; n < 64; ++n) { wl[n] = expf(wl[n] - mx); s += wl[n]; }
        float inv = 1.f / s;
        for (int n = 0; n < 64; ++n) wl[n] *= inv;
    }
    __syncthreads();
    for (int c = t; c < 512; c += 256) {
        float a = 0.f;
        for (int n = 0; n < 64; ++n) a += wl[n] * attm[((size_t)b * 64 + n) * 512 + c];
        mv[c] = a;
    }
    __syncthreads();
    if (t < 100) {
        const float* wr = w_fc + (size_t)t * 1024;
        float a = b_fc[t];
        for (int c = 0; c < 512; ++c) a += enc[c] * wr[c] + mv[c] * wr[512 + c];
        out[b * 100 + t] = a;
    }
}

// ---------------------------------------------------------------------------
extern "C" void kernel_launch(void* const* d_in, const int* in_sizes, int n_in,
                              void* d_out, int out_size, void* d_ws, size_t ws_size,
                              hipStream_t stream) {
    (void)in_sizes; (void)n_in; (void)out_size; (void)ws_size;
    const float* feat_b = (const float*)d_in[0];   // [8,512,10,10]
    const float* feat_m = (const float*)d_in[1];   // [64,512,10,10]
    const float* w1x1   = (const float*)d_in[2];   // [64,512]
    const float* gamma  = (const float*)d_in[3];   // [64]
    const float* beta   = (const float*)d_in[4];   // [64]
    const float* w1a    = (const float*)d_in[5];   // [16,1,3,3]
    const float* w1b    = (const float*)d_in[6];   // [16,16,3,3]
    const float* w2a    = (const float*)d_in[7];   // [1,16,3,3]
    const float* w2b    = (const float*)d_in[8];   // [1,1,3,3]
    const float* w_fc   = (const float*)d_in[9];   // [100,1024]
    const float* b_fc   = (const float*)d_in[10];  // [100]
    float* out = (float*)d_out;                    // [8,100]

    // ---- workspace layout (all 256B aligned; total ~52 MB) ----
    char* ws = (char*)d_ws;
    size_t off = 0;
    auto take = [&](size_t bytes) { char* p = ws + off; off = (off + bytes + 255) & ~(size_t)255; return p; };
    __bf16* fbp  = (__bf16*)take((size_t)8  * CP * 64 * 2);       // proj(feat_b) [b][pos][r]
    __bf16* fmp  = (__bf16*)take((size_t)64 * CP * 64 * 2);       // proj(feat_m) [n][pos][r]
    float*  corr = (float*) take((size_t)512 * CPP * 4);          // padded corr / CCA out
    __bf16* hbuf = (__bf16*)take((size_t)64 * 16 * 100 * CP * 2); // block-1 out, one chunk
    __bf16* W1bf = (__bf16*)take((size_t)16 * 96 * 2);
    float*  W2   = (float*) take((size_t)16 * 81 * 4);
    float*  wsum = (float*) take((size_t)64 * 4);
    float*  attn = (float*) take((size_t)8 * 64 * 100 * 4);
    float*  attm = (float*) take((size_t)8 * 64 * 512 * 4);

    // zero the padded bf16 projection buffers (pads must be 0 for WMMA tiles)
    int wz0 = 8 * CP * 64 * 2 / 4, wz1 = 64 * CP * 64 * 2 / 4;
    k_zero<<<(wz0 + 255) / 256, 256, 0, stream>>>((unsigned*)fbp, wz0);
    k_zero<<<(wz1 + 255) / 256, 256, 0, stream>>>((unsigned*)fmp, wz1);

    k_weights<<<1, 256, 0, stream>>>(w1x1, w1a, w1b, w2a, w2b, wsum, W1bf, W2);

    k_proj<<<8 * 100, 64, 0, stream>>>(feat_b, w1x1, wsum, gamma, beta, fbp);
    k_proj<<<64 * 100, 64, 0, stream>>>(feat_m, w1x1, wsum, gamma, beta, fmp);

    k_corr<<<512, 256, 0, stream>>>(fmp, fbp, corr);

    // CCA: chunk bn by 64 so h (~23MB bf16) stays L2-resident between blocks.
    for (int ch = 0; ch < 8; ++ch) {
        k_cca1<<<64, 256, 0, stream>>>(corr, W1bf, hbuf, ch * 64);
        k_cca2<<<640, 256, 0, stream>>>(hbuf, W2, corr, ch * 64);
    }

    k_attn<<<512, 128, 0, stream>>>(corr, attn);
    k_attmem<<<512, 256, 0, stream>>>(attn, feat_m, attm);
    k_head<<<8, 256, 0, stream>>>(feat_b, attm, w_fc, b_fc, out);
}
